// GPTModel_20332375179729
// MI455X (gfx1250) — matrix operand order repaired and checked by reading, not compiled, not run on hardware
//
#include <hip/hip_runtime.h>
#include <hip/hip_bf16.h>

// ---- model constants (from reference) ----
#define Vv    50257
#define Cc    768
#define Hh    12
#define Ll    6
#define FFd   (4*Cc)
#define HDd   (Cc/Hh)     // 64
#define Bb    2
#define Tt    1024
#define NBT   (Bb*Tt)     // 2048

typedef __attribute__((ext_vector_type(16))) _Float16 v16h;
typedef __attribute__((ext_vector_type(8)))  float    v8f;

// ---- GEMM tiling ----
#define BM 128            // block rows
#define BN 64             // block cols
#define BK 32             // k-step (one WMMA K)
#define APITCH 40         // halves per LDS row (pad 8 -> rows stay 16B aligned)
#define BPITCH 40
#define NTHREADS 128      // 4 waves; each wave owns 32 rows x 64 cols = 8 accumulators

// Generic f32-in / f16-WMMA / f32-out GEMM, software-pipelined:
//   C[m,n] = sum_k A[m,k] * B'[k,n]   (B' = B or B^T per opB)
// batched head addressing: z -> (zo=z/Hmod, zi=z%Hmod); offset = zo*outer + zi*inner
// epilogue: +bias[n], relu, +res[m,n]
// All global loads are unconditional (indices clamped); OOB lanes only feed
// output elements the epilogue guards. Tiles are register-staged (all loads in
// flight at once), LDS is double-buffered (one barrier per k-step), and all
// 6 fragments (2 A + 4 B) are preloaded so the 8 WMMAs issue as one dense chain.
__global__ __launch_bounds__(NTHREADS)
void gemm_f16wmma(const float* __restrict__ A, int lda, long long aOuter, long long aInner,
                  const float* __restrict__ Bm, int ldb, long long bOuter, long long bInner, int opB,
                  float* __restrict__ Cm, int ldc, long long cOuter, long long cInner,
                  int M, int N, int K, int Hmod,
                  const float* __restrict__ bias, const float* __restrict__ res, int relu)
{
    __shared__ _Float16 As[2][BM * APITCH];     // [m][k]
    __shared__ _Float16 Bs[2][BN * BPITCH];     // [n][k]  (transposed tile)

    const int tid  = threadIdx.x;
    const int lane = tid & 31;
    const int wave = tid >> 5;                  // 0..3, owns rows [wave*32, wave*32+32)
    const int m0 = blockIdx.y * BM;
    const int n0 = blockIdx.x * BN;

    const int z  = blockIdx.z;
    const int zo = z / Hmod;
    const int zi = z - zo * Hmod;
    const long long aOff = (long long)zo * aOuter + (long long)zi * aInner;
    const long long bOff = (long long)zo * bOuter + (long long)zi * bInner;
    const long long cOff = (long long)zo * cOuter + (long long)zi * cInner;

    const int lm15 = lane & 15;
    const int lhi  = lane >> 4;                 // 0 or 1

    v8f acc[8];
    v8f zeroAcc = {};
#pragma unroll
    for (int j = 0; j < 8; ++j) acc[j] = zeroAcc;

    const bool bVec4N = (opB == 0) && ((ldb & 3) == 0) && ((N & (BN - 1)) == 0);

    // -------- register staging (all loads issued back-to-back) --------
    float4 aR[8];
    float4 bR4[4];
    float  bRs[16];

    auto loadTile = [&](int k0) {
#pragma unroll
        for (int i = 0; i < 8; ++i) {           // A: 128x32, float4 slots
            int idx = tid + i * NTHREADS;
            int r = idx >> 3, c4 = idx & 7;
            int gm = m0 + r; gm = gm < M ? gm : M - 1;
            aR[i] = *(const float4*)(A + aOff + (long long)gm * lda + (k0 + c4 * 4));
        }
        if (opB) {                              // B rows are n, contiguous in k
#pragma unroll
            for (int i = 0; i < 4; ++i) {
                int idx = tid + i * NTHREADS;
                int n = idx >> 3, c4 = idx & 7;
                int gn = n0 + n; gn = gn < N ? gn : N - 1;
                bR4[i] = *(const float4*)(Bm + bOff + (long long)gn * ldb + (k0 + c4 * 4));
            }
        } else if (bVec4N) {                    // rows are k, contiguous+aligned in n
#pragma unroll
            for (int i = 0; i < 4; ++i) {
                int idx = tid + i * NTHREADS;
                int nq = idx & 15, kx = idx >> 4;
                bR4[i] = *(const float4*)(Bm + bOff + (long long)(k0 + kx) * ldb + (n0 + nq * 4));
            }
        } else {                                // general: coalesced b32 along n
#pragma unroll
            for (int i = 0; i < 16; ++i) {
                int idx = tid + i * NTHREADS;
                int n = idx & (BN - 1), kx = idx >> 6;
                int gn = n0 + n; gn = gn < N ? gn : N - 1;
                bRs[i] = Bm[bOff + (long long)(k0 + kx) * ldb + gn];
            }
        }
    };

    auto storeTile = [&](int buf) {
        _Float16* Ad = As[buf];
        _Float16* Bd = Bs[buf];
#pragma unroll
        for (int i = 0; i < 8; ++i) {
            int idx = tid + i * NTHREADS;
            int r = idx >> 3, c4 = idx & 7;
            _Float16* dst = &Ad[r * APITCH + c4 * 4];
            dst[0] = (_Float16)aR[i].x; dst[1] = (_Float16)aR[i].y;
            dst[2] = (_Float16)aR[i].z; dst[3] = (_Float16)aR[i].w;
        }
        if (opB) {
#pragma unroll
            for (int i = 0; i < 4; ++i) {
                int idx = tid + i * NTHREADS;
                int n = idx >> 3, c4 = idx & 7;
                _Float16* dst = &Bd[n * BPITCH + c4 * 4];
                dst[0] = (_Float16)bR4[i].x; dst[1] = (_Float16)bR4[i].y;
                dst[2] = (_Float16)bR4[i].z; dst[3] = (_Float16)bR4[i].w;
            }
        } else if (bVec4N) {
#pragma unroll
            for (int i = 0; i < 4; ++i) {
                int idx = tid + i * NTHREADS;
                int nq = idx & 15, kx = idx >> 4;
                Bd[(nq * 4 + 0) * BPITCH + kx] = (_Float16)bR4[i].x;
                Bd[(nq * 4 + 1) * BPITCH + kx] = (_Float16)bR4[i].y;
                Bd[(nq * 4 + 2) * BPITCH + kx] = (_Float16)bR4[i].z;
                Bd[(nq * 4 + 3) * BPITCH + kx] = (_Float16)bR4[i].w;
            }
        } else {
#pragma unroll
            for (int i = 0; i < 16; ++i) {
                int idx = tid + i * NTHREADS;
                int n = idx & (BN - 1), kx = idx >> 6;
                Bd[n * BPITCH + kx] = (_Float16)bRs[i];
            }
        }
    };

    // -------- pipelined main loop --------
    const int ktiles = K / BK;
    loadTile(0);
    storeTile(0);

    for (int kt = 0; kt < ktiles; ++kt) {
        __syncthreads();                        // buf (kt&1) ready for all waves
        const int cur = kt & 1;
        const bool more = (kt + 1) < ktiles;
        if (more) loadTile((kt + 1) * BK);      // overlap with compute below

        // ---- preload ALL fragments (2 A + 4 B) so WMMAs issue densely ----
        // A: lane L holds M=L&15; halves 0-7 -> K=(L>>4)*8.., 8-15 -> +16
        const _Float16* Ap0 = &As[cur][(wave * 32 + lm15) * APITCH];
        const _Float16* Ap1 = Ap0 + 16 * APITCH;
        const int kb = lhi * 8;
        v16h a0, a1;
#pragma unroll
        for (int i2 = 0; i2 < 8; ++i2) {
            a0[i2]     = Ap0[kb + i2];
            a0[i2 + 8] = Ap0[16 + kb + i2];
            a1[i2]     = Ap1[kb + i2];
            a1[i2 + 8] = Ap1[16 + kb + i2];
        }
        // B: lane L holds N=L&15; halves 0-7 -> K=(L>>4)*8.., 8-15 -> +16 (transposed LDS tile)
        v16h bf[4];
#pragma unroll
        for (int j = 0; j < 4; ++j) {
            const _Float16* Bp = &Bs[cur][(j * 16 + lm15) * BPITCH + lhi * 8];
#pragma unroll
            for (int i2 = 0; i2 < 16; ++i2) bf[j][i2] = Bp[i2 + (i2 & 8)];
        }
        // ---- 8 WMMAs, back-to-back ----
#pragma unroll
        for (int j = 0; j < 4; ++j) {
            acc[j] = __builtin_amdgcn_wmma_f32_16x16x32_f16(
                false, a0, false, bf[j], (short)0, acc[j], false, false);
            acc[j + 4] = __builtin_amdgcn_wmma_f32_16x16x32_f16(
                false, a1, false, bf[j], (short)0, acc[j + 4], false, false);
        }

        if (more) storeTile(1 - cur);           // convert + stage next tile
    }

    // ---- epilogue / store: VGPR r -> M = r + (L>>4)*8, N = L&15 ----
#pragma unroll
    for (int f = 0; f < 2; ++f) {
#pragma unroll
        for (int j = 0; j < 4; ++j) {
            int gn = n0 + j * 16 + lm15;
            if (gn >= N) continue;
            float bv = bias ? bias[gn] : 0.0f;
            int gmb = m0 + wave * 32 + f * 16 + lhi * 8;
#pragma unroll
            for (int r = 0; r < 8; ++r) {
                int gm = gmb + r;
                if (gm >= M) continue;
                float out = acc[f * 4 + j][r] + bv;
                if (relu) out = fmaxf(out, 0.0f);
                if (res) out += res[cOff + (long long)gm * ldc + gn];
                Cm[cOff + (long long)gm * ldc + gn] = out;
            }
        }
    }
}

// ---- embedding: x[b,t,:] = wte[idx[b,t],:] + wpe[t,:] ----
__global__ void embed_kernel(const int* __restrict__ idx, const float* __restrict__ wte,
                             const float* __restrict__ wpe, float* __restrict__ x, int total)
{
    int i = blockIdx.x * blockDim.x + threadIdx.x;
    if (i >= total) return;
    int row = i / Cc, c = i - row * Cc;
    int t = row & (Tt - 1);
    int tok = idx[row];
    x[i] = wte[(long long)tok * Cc + c] + wpe[(long long)t * Cc + c];
}

// ---- layernorm over rows of length n ----
__global__ __launch_bounds__(256)
void layernorm_rows(const float* __restrict__ x, const float* __restrict__ g,
                    const float* __restrict__ b, float* __restrict__ y, int n)
{
    __shared__ float sh[256];
    int row = blockIdx.x, tid = threadIdx.x;
    const float* px = x + (long long)row * n;
    float s = 0.0f, sq = 0.0f;
    for (int c = tid; c < n; c += 256) { float v = px[c]; s += v; sq += v * v; }
    sh[tid] = s; __syncthreads();
    for (int st = 128; st > 0; st >>= 1) { if (tid < st) sh[tid] += sh[tid + st]; __syncthreads(); }
    float mean = sh[0] / n; __syncthreads();
    sh[tid] = sq; __syncthreads();
    for (int st = 128; st > 0; st >>= 1) { if (tid < st) sh[tid] += sh[tid + st]; __syncthreads(); }
    float var = sh[0] / n - mean * mean;
    float rstd = rsqrtf(var + 1e-5f);
    float* py = y + (long long)row * n;
    for (int c = tid; c < n; c += 256) py[c] = (px[c] - mean) * rstd * g[c] + b[c];
}

// ---- causal softmax, in place; row = z*T + t ; scale = C**-0.5 (faithful) ----
__global__ __launch_bounds__(256)
void softmax_causal(float* __restrict__ sc, int T, float scale)
{
    __shared__ float sh[256];
    int row = blockIdx.x, tid = threadIdx.x;
    int t = row & (T - 1);
    float* p = sc + (long long)row * T;
    int len = t + 1;
    float mx = -3.4e38f;
    for (int s = tid; s < len; s += 256) mx = fmaxf(mx, p[s] * scale);
    sh[tid] = mx; __syncthreads();
    for (int st = 128; st > 0; st >>= 1) { if (tid < st) sh[tid] = fmaxf(sh[tid], sh[tid + st]); __syncthreads(); }
    mx = sh[0]; __syncthreads();
    float sum = 0.0f;
    for (int s = tid; s < len; s += 256) sum += __expf(p[s] * scale - mx);
    sh[tid] = sum; __syncthreads();
    for (int st = 128; st > 0; st >>= 1) { if (tid < st) sh[tid] += sh[tid + st]; __syncthreads(); }
    float inv = 1.0f / sh[0];
    for (int s = tid; s < T; s += 256)
        p[s] = (s < len) ? __expf(p[s] * scale - mx) * inv : 0.0f;
}

// ---- per-row NLL: rl[m] = logsumexp(logits[m,:]) - logits[m, target[m]] ----
__global__ __launch_bounds__(256)
void loss_rows(const float* __restrict__ logits, const int* __restrict__ tgt, float* __restrict__ rl)
{
    __shared__ float sh[256];
    int m = blockIdx.x, tid = threadIdx.x;
    const float* p = logits + (long long)m * Vv;
    float mx = -3.4e38f;
    for (int j = tid; j < Vv; j += 256) mx = fmaxf(mx, p[j]);
    sh[tid] = mx; __syncthreads();
    for (int st = 128; st > 0; st >>= 1) { if (tid < st) sh[tid] = fmaxf(sh[tid], sh[tid + st]); __syncthreads(); }
    mx = sh[0]; __syncthreads();
    float sum = 0.0f;
    for (int j = tid; j < Vv; j += 256) sum += __expf(p[j] - mx);
    sh[tid] = sum; __syncthreads();
    for (int st = 128; st > 0; st >>= 1) { if (tid < st) sh[tid] += sh[tid + st]; __syncthreads(); }
    if (tid == 0) rl[m] = logf(sh[0]) + mx - p[tgt[m]];
}

__global__ __launch_bounds__(256)
void loss_final(const float* __restrict__ rl, float* __restrict__ out, int n)
{
    __shared__ float sh[256];
    int tid = threadIdx.x;
    float s = 0.0f;
    for (int j = tid; j < n; j += 256) s += rl[j];
    sh[tid] = s; __syncthreads();
    for (int st = 128; st > 0; st >>= 1) { if (tid < st) sh[tid] += sh[tid + st]; __syncthreads(); }
    if (tid == 0) out[0] = sh[0] / n;
}

extern "C" void kernel_launch(void* const* d_in, const int* in_sizes, int n_in,
                              void* d_out, int out_size, void* d_ws, size_t ws_size,
                              hipStream_t stream)
{
    (void)in_sizes; (void)n_in; (void)out_size; (void)ws_size;
    const int*   idx     = (const int*)d_in[0];
    const int*   targets = (const int*)d_in[1];
    const float* wte     = (const float*)d_in[2];
    const float* wpe     = (const float*)d_in[3];
    const float* ln1_g   = (const float*)d_in[4];
    const float* ln1_b   = (const float*)d_in[5];
    const float* wq      = (const float*)d_in[6];
    const float* wk      = (const float*)d_in[7];
    const float* wv      = (const float*)d_in[8];
    const float* apw     = (const float*)d_in[9];
    const float* apb     = (const float*)d_in[10];
    const float* ln2_g   = (const float*)d_in[11];
    const float* ln2_b   = (const float*)d_in[12];
    const float* fcw     = (const float*)d_in[13];
    const float* fcb     = (const float*)d_in[14];
    const float* prw     = (const float*)d_in[15];
    const float* prb     = (const float*)d_in[16];
    const float* lnfg    = (const float*)d_in[17];
    const float* lnfb    = (const float*)d_in[18];
    const float* lmw     = (const float*)d_in[19];

    float* logits = (float*)d_out;                       // [NBT, V]
    float* loss   = logits + (long long)NBT * Vv;        // scalar

    float* ws = (float*)d_ws;
    long long o = 0;
    float* x   = ws + o; o += (long long)NBT * Cc;       // activations
    float* hb  = ws + o; o += (long long)NBT * Cc;       // layernorm out
    float* q   = ws + o; o += (long long)NBT * Cc;
    float* kk  = ws + o; o += (long long)NBT * Cc;
    float* vvb = ws + o; o += (long long)NBT * Cc;
    float* ao  = ws + o; o += (long long)NBT * Cc;       // attention output
    float* ff  = ws + o; o += (long long)NBT * FFd;      // MLP hidden
    float* sc  = ws + o; o += (long long)Bb * Hh * Tt * Tt;  // attention scores
    float* rl  = ws + o; o += NBT;                       // per-row losses

    auto gemm = [&](const float* A, int lda, long long aO, long long aI,
                    const float* Bp, int ldb, long long bO, long long bI, int opB,
                    float* Cp, int ldc, long long cO, long long cI,
                    int M, int N, int K, int batch, int Hmod,
                    const float* bias, const float* res, int relu) {
        dim3 g((N + BN - 1) / BN, (M + BM - 1) / BM, batch);
        gemm_f16wmma<<<g, dim3(NTHREADS), 0, stream>>>(A, lda, aO, aI, Bp, ldb, bO, bI, opB,
                                                       Cp, ldc, cO, cI, M, N, K, Hmod,
                                                       bias, res, relu);
    };

    // x = wte[idx] + wpe
    {
        int total = NBT * Cc;
        embed_kernel<<<(total + 255) / 256, 256, 0, stream>>>(idx, wte, wpe, x, total);
    }

    const float scale = 1.0f / sqrtf((float)Cc);   // faithful: n_embd**-0.5

    for (int l = 0; l < Ll; ++l) {
        const float* wql = wq  + (long long)l * Cc * Cc;
        const float* wkl = wk  + (long long)l * Cc * Cc;
        const float* wvl = wv  + (long long)l * Cc * Cc;
        const float* apl = apw + (long long)l * Cc * Cc;
        const float* fwl = fcw + (long long)l * Cc * FFd;
        const float* pwl = prw + (long long)l * FFd * Cc;

        // h = LN1(x)
        layernorm_rows<<<NBT, 256, 0, stream>>>(x, ln1_g + l * Cc, ln1_b + l * Cc, hb, Cc);
        // q,k,v = h @ W
        gemm(hb, Cc, 0, 0, wql, Cc, 0, 0, 0, q,   Cc, 0, 0, NBT, Cc, Cc, 1, 1, nullptr, nullptr, 0);
        gemm(hb, Cc, 0, 0, wkl, Cc, 0, 0, 0, kk,  Cc, 0, 0, NBT, Cc, Cc, 1, 1, nullptr, nullptr, 0);
        gemm(hb, Cc, 0, 0, wvl, Cc, 0, 0, 0, vvb, Cc, 0, 0, NBT, Cc, Cc, 1, 1, nullptr, nullptr, 0);
        // scores[b,h] = Q_head @ K_head^T   (batched over B*H, head offset = h*HD)
        gemm(q,  Cc, (long long)Tt * Cc, HDd,
             kk, Cc, (long long)Tt * Cc, HDd, /*opB=T*/1,
             sc, Tt, (long long)Hh * Tt * Tt, (long long)Tt * Tt,
             Tt, Tt, HDd, Bb * Hh, Hh, nullptr, nullptr, 0);
        // causal softmax with C**-0.5 scale
        softmax_causal<<<Bb * Hh * Tt, 256, 0, stream>>>(sc, Tt, scale);
        // attn = W @ V_head   (batched)
        gemm(sc,  Tt, (long long)Hh * Tt * Tt, (long long)Tt * Tt,
             vvb, Cc, (long long)Tt * Cc, HDd, /*opB=N*/0,
             ao,  Cc, (long long)Tt * Cc, HDd,
             Tt, HDd, Tt, Bb * Hh, Hh, nullptr, nullptr, 0);
        // x = x + attn @ apw + apb
        gemm(ao, Cc, 0, 0, apl, Cc, 0, 0, 0,
             x, Cc, 0, 0, NBT, Cc, Cc, 1, 1, apb + (long long)l * Cc, x, 0);
        // h = LN2(x)
        layernorm_rows<<<NBT, 256, 0, stream>>>(x, ln2_g + l * Cc, ln2_b + l * Cc, hb, Cc);
        // ff = relu(h @ fc_w + fc_b)
        gemm(hb, Cc, 0, 0, fwl, FFd, 0, 0, 0,
             ff, FFd, 0, 0, NBT, FFd, Cc, 1, 1, fcb + (long long)l * FFd, nullptr, 1);
        // x = x + ff @ pr_w + pr_b
        gemm(ff, FFd, 0, 0, pwl, Cc, 0, 0, 0,
             x, Cc, 0, 0, NBT, Cc, FFd, 1, 1, prb + (long long)l * Cc, x, 0);
    }

    // final LN + LM head
    layernorm_rows<<<NBT, 256, 0, stream>>>(x, lnfg, lnfb, hb, Cc);
    gemm(hb, Cc, 0, 0, lmw, Vv, 0, 0, 0,
         logits, Vv, 0, 0, NBT, Vv, Cc, 1, 1, nullptr, nullptr, 0);

    // loss = mean(logsumexp - logit[target])
    loss_rows<<<NBT, 256, 0, stream>>>(logits, targets, rl);
    loss_final<<<1, 256, 0, stream>>>(rl, loss, NBT);
}
